// SelfAttentionExtractor_21122649162109
// MI455X (gfx1250) — matrix operand order repaired and checked by reading, not compiled, not run on hardware
//
#include <hip/hip_runtime.h>

typedef __attribute__((ext_vector_type(16))) _Float16 v16h;
typedef __attribute__((ext_vector_type(8)))  _Float16 v8h;
typedef __attribute__((ext_vector_type(8)))  float    v8f;

#define S      36     // tokens per batch element (6 goals * 6 objects)
#define H      256    // hidden
#define NH     8      // heads
#define HD     32     // head dim
#define NBLK   4      // attention blocks
#define NWAVES 8      // 256 threads / wave32

// Pre-packed weight workspace layout (all f16, WMMA B-fragment order):
//   ew1p: 33x256 padded to K=64 -> 2 kc * 16 nt blocks * 512 halves = 16384
//   then 25 HxH matrices (8 kc * 16 nt blocks * 512 halves = 65536 each):
//   [ew2][Wq*4][Wk*4][Wv*4][Wo*4][fw1*4][fw2*4]
#define EW1P_HALVES 16384
#define MAT_HALVES  65536
#define NMATS       25
#define WS_TOTAL    (EW1P_HALVES + NMATS * MAT_HALVES)

struct Params {
  const float* robot;          // [B,47]
  const float* objs;           // [B,6,16]
  const unsigned char* masks;  // [B,6] bool (True = masked key)
  const float *ew1, *eb1, *ew2, *eb2, *elg, *elb;
  const float *Wq, *bq, *Wk, *bk, *Wv, *bv, *Wo, *bo;
  const float *ln1g, *ln1b, *fw1, *fb1, *fw2, *fb2, *ln2g, *ln2b;
  float* out;                  // [B,H]
};

__device__ inline v8f wmma_f16(v16h a, v16h b, v8f c) {
  return __builtin_amdgcn_wmma_f32_16x16x32_f16(false, a, false, b, (short)0, c,
                                                false, false);
}

// ---------------------------------------------------------------------------
// Pre-pack kernel: f32 weights -> f16 fragment-ordered blocks in d_ws.
// Block (kc,nt) of a KxN matrix stores, for lane l and slot j:
//   k = kc*32 + ((l&16)?16:0) + j ,  n = nt*16 + (l&15)
// at ws[((nt*nkc + kc) << 9) + l*16 + j].
// ---------------------------------------------------------------------------
__global__ __launch_bounds__(256) void prepack_kernel(Params p, _Float16* ws) {
  int idx = blockIdx.x * 256 + threadIdx.x;
  if (idx >= WS_TOTAL) return;
  if (idx < EW1P_HALVES) {  // embed layer-1, K padded 33 -> 64 (nkc = 2)
    int r = idx;
    int block = r >> 9, lane = (r >> 4) & 31, j = r & 15;
    int nt = block >> 1, kc = block & 1;
    int k = kc * 32 + ((lane & 16) ? 16 : 0) + j;
    int n = nt * 16 + (lane & 15);
    ws[idx] = (_Float16)((k < 33) ? p.ew1[k * H + n] : 0.f);
    return;
  }
  int d = idx - EW1P_HALVES;
  int m = d >> 16, r = d & (MAT_HALVES - 1);
  int block = r >> 9, lane = (r >> 4) & 31, j = r & 15;
  int nt = block >> 3, kc = block & 7;
  int k = kc * 32 + ((lane & 16) ? 16 : 0) + j;
  int n = nt * 16 + (lane & 15);
  const float* Wm;
  if (m == 0)       Wm = p.ew2;
  else if (m < 5)   Wm = p.Wq + (m - 1) * MAT_HALVES;
  else if (m < 9)   Wm = p.Wk + (m - 5) * MAT_HALVES;
  else if (m < 13)  Wm = p.Wv + (m - 9) * MAT_HALVES;
  else if (m < 17)  Wm = p.Wo + (m - 13) * MAT_HALVES;
  else if (m < 21)  Wm = p.fw1 + (m - 17) * MAT_HALVES;
  else              Wm = p.fw2 + (m - 21) * MAT_HALVES;
  ws[idx] = (_Float16)Wm[k * H + n];
}

// ---------------------------------------------------------------------------
// Fragment packers (all loads UNGUARDED; see per-site safety comments)
// ---------------------------------------------------------------------------

// A fragment: 16x32 f16 tile at (row mt*16, col kOff) of row-major LDS buffer.
// Overreads land only in D rows >= 36, which are never stored.
__device__ inline v16h pack_a(const _Float16* src, int ld, int mt, int kOff,
                              int lane) {
  int row = mt * 16 + (lane & 15);
  int kb = kOff + ((lane & 16) ? 16 : 0);
  const v8h* pp = (const v8h*)(src + row * ld + kb);
  v8h lo = pp[0];
  v8h hi = pp[1];  // +8 halves
  v16h a;
#pragma unroll
  for (int j = 0; j < 8; ++j) {
    a[j] = lo[j];
    a[8 + j] = hi[j];
  }
  return a;
}

// B fragment from pre-packed global weights: 2x global_load_b128 per lane.
__device__ inline v16h pack_b_p(const _Float16* Wp, int kc, int nt, int nkc,
                                int lane) {
  const v8h* pp = (const v8h*)(Wp + (((nt * nkc + kc) << 9) + lane * 16));
  v8h lo = pp[0];
  v8h hi = pp[1];
  v16h b;
#pragma unroll
  for (int j = 0; j < 8; ++j) {
    b[j] = lo[j];
    b[8 + j] = hi[j];
  }
  return b;
}

// B fragment from row-major LDS f16 matrix B[k][n] (strided; K-overrun rows
// are multiplied by zero probabilities).
__device__ inline v16h pack_b_l(const _Float16* Bm, int ldn, int kOff, int nOff,
                                int lane) {
  int n = nOff + (lane & 15);
  int kb = kOff + ((lane & 16) ? 16 : 0);
  v16h b;
#pragma unroll
  for (int j = 0; j < 16; ++j) b[j] = Bm[(kb + j) * ldn + n];
  return b;
}

// B fragment = transpose of row-major LDS matrix: B[k][n] = Km[n][k].
// Garbage N columns >= 36 produce score columns the softmax discards.
__device__ inline v16h pack_bt_l(const _Float16* Km, int ldk, int nOff,
                                 int lane) {
  int n = nOff + (lane & 15);
  int kb = (lane & 16) ? 16 : 0;
  const v8h* pp = (const v8h*)(Km + n * ldk + kb);
  v8h lo = pp[0];
  v8h hi = pp[1];
  v16h b;
#pragma unroll
  for (int j = 0; j < 8; ++j) {
    b[j] = lo[j];
    b[8 + j] = hi[j];
  }
  return b;
}

// dst[36,H] = act( src[36,srcLd] @ Wp + bias (+ resid) ), output stored f16.
__device__ inline void gemm_HxH(const _Float16* src, int srcLd, int nkc,
                                const _Float16* Wp, const float* bias,
                                _Float16* dst, const _Float16* resid, int relu,
                                int wave, int lane) {
  for (int t = wave; t < 3 * 16; t += NWAVES) {  // wave-uniform tile loop
    int mt = t >> 4, nt = t & 15;
    v8f c = {0.f, 0.f, 0.f, 0.f, 0.f, 0.f, 0.f, 0.f};
    for (int kc = 0; kc < nkc; ++kc) {
      v16h a = pack_a(src, srcLd, mt, kc * 32, lane);
      v16h b = pack_b_p(Wp, kc, nt, nkc, lane);
      c = wmma_f16(a, b, c);
    }
    int col = nt * 16 + (lane & 15);
    float bb = bias[col];
    if (mt < 2) {  // rows all valid: unconditional stores
      int rb = mt * 16 + ((lane & 16) ? 8 : 0);
#pragma unroll
      for (int r = 0; r < 8; ++r) {
        int row = rb + r;
        float v = c[r] + bb;
        if (resid) v += (float)resid[row * H + col];
        if (relu) v = fmaxf(v, 0.f);
        dst[row * H + col] = (_Float16)v;
      }
    } else if (!(lane & 16)) {  // mt==2: only lanes 0-15, rows 32..35
#pragma unroll
      for (int r = 0; r < 4; ++r) {
        int row = 32 + r;
        float v = c[r] + bb;
        if (resid) v += (float)resid[row * H + col];
        if (relu) v = fmaxf(v, 0.f);
        dst[row * H + col] = (_Float16)v;
      }
    }
  }
}

// Row-wise LayerNorm over H=256 features, one wave per row.
__device__ inline void layer_norm(_Float16* xb, const float* g, const float* bt,
                                  int wave, int lane) {
  for (int r = wave; r < S; r += NWAVES) {
    float vals[8];
    float s = 0.f;
#pragma unroll
    for (int i = 0; i < 8; ++i) {
      vals[i] = (float)xb[r * H + lane + 32 * i];
      s += vals[i];
    }
#pragma unroll
    for (int off = 16; off >= 1; off >>= 1) s += __shfl_xor(s, off, 32);
    float mu = s * (1.f / 256.f);
    float vs = 0.f;
#pragma unroll
    for (int i = 0; i < 8; ++i) {
      float d = vals[i] - mu;
      vs += d * d;
    }
#pragma unroll
    for (int off = 16; off >= 1; off >>= 1) vs += __shfl_xor(vs, off, 32);
    float inv = rsqrtf(vs * (1.f / 256.f) + 1e-5f);
#pragma unroll
    for (int i = 0; i < 8; ++i) {
      int col = lane + 32 * i;
      xb[r * H + col] = (_Float16)((vals[i] - mu) * inv * g[col] + bt[col]);
    }
  }
}

__global__ __launch_bounds__(256) void attn_extractor_kernel(
    Params p, const _Float16* ws) {
  __shared__ __align__(16) _Float16 bufA[S * H];  // activations x (18 KB)
  __shared__ __align__(16) _Float16 bufB[S * H];  // attn concat / ffn hidden
  __shared__ __align__(16) _Float16 bufC[12288];  // LN1 out / scratch (24 KB)
  __shared__ float maskbias[8];                   // 0 or -1e9 per object slot

  const int b = blockIdx.x;
  const int tid = threadIdx.x;
  const int wave = tid >> 5;
  const int lane = tid & 31;

  const _Float16* ew1p = ws;
  const _Float16* mats = ws + EW1P_HALVES;

  // Attention scratch inside bufC (time-disjoint with its activation role).
  // All regions padded to 48 rows so stores need no guards.
  _Float16* qh = bufC;                   // [48][32]  (3 KB)
  _Float16* kh = bufC + 48 * HD;         // [48][32]
  _Float16* vh = bufC + 2 * 48 * HD;     // [48][32]
  float* sc = (float*)(bufC + 3 * 48 * HD);               // [48][48] f32
  _Float16* pr = (_Float16*)((char*)bufC + 9216 + 9216);  // [36+][64] f16

  // ---- build tokens [48][64] f16 into bufC (embed phase only) ----
  _Float16* tokA = bufC;  // 48*64 halves, zero-padded rows/cols
  for (int idx = tid; idx < 48 * 64; idx += 256) tokA[idx] = (_Float16)0.f;
  if (tid < 6) maskbias[tid] = p.masks[b * 6 + tid] ? -1e9f : 0.f;
  if (tid >= 6 && tid < 8) maskbias[tid] = 0.f;
  __syncthreads();
  for (int idx = tid; idx < S * 33; idx += 256) {
    int row = idx / 33, c = idx - row * 33;
    int g = row / 6, o = row - g * 6;
    float v;
    if (c < 11)      v = p.robot[b * 47 + c];
    else if (c < 14) v = p.robot[b * 47 + 11 + g * 3 + (c - 11)];
    else if (c < 17) v = p.robot[b * 47 + 29 + g * 3 + (c - 14)];
    else             v = p.objs[(b * 6 + o) * 16 + (c - 17)];
    tokA[row * 64 + c] = (_Float16)v;
  }
  __syncthreads();

  // ---- embed layer1 (WMMA, K=64): bufB = relu(tokA @ ew1p + eb1) ----
  gemm_HxH(tokA, 64, 2, ew1p, p.eb1, bufB, nullptr, 1, wave, lane);
  __syncthreads();

  // ---- embed layer2 (WMMA) + LN -> bufA ----
  gemm_HxH(bufB, H, 8, mats, p.eb2, bufA, nullptr, 0, wave, lane);
  __syncthreads();
  layer_norm(bufA, p.elg, p.elb, wave, lane);
  __syncthreads();

  const float scale = 0.17677669529663687f;  // 1/sqrt(32)

  for (int i = 0; i < NBLK; ++i) {
    const _Float16* Wqp = mats + (1 + i) * MAT_HALVES;
    const _Float16* Wkp = mats + (5 + i) * MAT_HALVES;
    const _Float16* Wvp = mats + (9 + i) * MAT_HALVES;
    const float* bqi = p.bq + i * H;
    const float* bki = p.bk + i * H;
    const float* bvi = p.bv + i * H;

    for (int h = 0; h < NH; ++h) {
      // Q/K/V head projections: 18 tiles (3 mats * 3 M-tiles * 2 N-tiles).
      // dst buffers are 48-row padded -> unconditional stores.
      for (int t = wave; t < 18; t += NWAVES) {
        int which = t / 6, tt = t - which * 6;
        int mt = tt >> 1, nt = tt & 1;
        const _Float16* Wp = (which == 0) ? Wqp : (which == 1) ? Wkp : Wvp;
        const float* bs = (which == 0) ? bqi : (which == 1) ? bki : bvi;
        _Float16* dst = (which == 0) ? qh : (which == 1) ? kh : vh;
        v8f c = {0.f, 0.f, 0.f, 0.f, 0.f, 0.f, 0.f, 0.f};
        for (int kc = 0; kc < 8; ++kc) {
          v16h a = pack_a(bufA, H, mt, kc * 32, lane);
          v16h bf = pack_b_p(Wp, kc, h * 2 + nt, 8, lane);
          c = wmma_f16(a, bf, c);
        }
        int col = nt * 16 + (lane & 15);
        int rb = mt * 16 + ((lane & 16) ? 8 : 0);
        float bb = bs[h * HD + col];
#pragma unroll
        for (int r = 0; r < 8; ++r)
          dst[(rb + r) * HD + col] = (_Float16)(c[r] + bb);
      }
      __syncthreads();

      // scores: sc[q][k] = (qh @ kh^T) * scale  (K=32, one WMMA per tile);
      // sc is 48x48 padded -> unconditional stores.
      for (int t = wave; t < 9; t += NWAVES) {
        int mt = t / 3, nt = t - mt * 3;
        v16h a = pack_a(qh, HD, mt, 0, lane);
        v16h bf = pack_bt_l(kh, HD, nt * 16, lane);
        v8f c = {0.f, 0.f, 0.f, 0.f, 0.f, 0.f, 0.f, 0.f};
        c = wmma_f16(a, bf, c);
        int col = nt * 16 + (lane & 15);
        int rb = mt * 16 + ((lane & 16) ? 8 : 0);
#pragma unroll
        for (int r = 0; r < 8; ++r) sc[(rb + r) * 48 + col] = c[r] * scale;
      }
      __syncthreads();

      // masked softmax per query row (one wave per row). pr has stride 64 and
      // every lane writes key slots {lane, 32+lane}: columns 36..63 get exact
      // zeros so the P@V K-overrun contributes nothing.
      for (int q = wave; q < S; q += NWAVES) {
        int k0 = lane, k1 = 32 + lane;
        float s0 = sc[q * 48 + k0] + maskbias[k0 % 6];
        float s1 = (k1 < S) ? (sc[q * 48 + k1] + maskbias[k1 % 6]) : -INFINITY;
        float mx = fmaxf(s0, s1);
#pragma unroll
        for (int off = 16; off >= 1; off >>= 1)
          mx = fmaxf(mx, __shfl_xor(mx, off, 32));
        float e0 = __expf(s0 - mx);
        float e1 = (k1 < S) ? __expf(s1 - mx) : 0.f;
        float sm = e0 + e1;
#pragma unroll
        for (int off = 16; off >= 1; off >>= 1) sm += __shfl_xor(sm, off, 32);
        float inv = 1.f / sm;
        pr[q * 64 + k0] = (_Float16)(e0 * inv);
        pr[q * 64 + k1] = (_Float16)(e1 * inv);  // zero for k1 >= 36
      }
      __syncthreads();

      // attn_h = P[36,64] @ V[64->36,32] -> bufB[:, h*32 : h*32+32]
      for (int t = wave; t < 6; t += NWAVES) {
        int mt = t >> 1, nt = t & 1;
        v8f c = {0.f, 0.f, 0.f, 0.f, 0.f, 0.f, 0.f, 0.f};
        for (int kc = 0; kc < 2; ++kc) {
          v16h a = pack_a(pr, 64, mt, kc * 32, lane);
          v16h bf = pack_b_l(vh, HD, kc * 32, nt * 16, lane);
          c = wmma_f16(a, bf, c);
        }
        int col = nt * 16 + (lane & 15);
        if (mt < 2) {
          int rb = mt * 16 + ((lane & 16) ? 8 : 0);
#pragma unroll
          for (int r = 0; r < 8; ++r)
            bufB[(rb + r) * H + h * HD + col] = (_Float16)c[r];
        } else if (!(lane & 16)) {
#pragma unroll
          for (int r = 0; r < 4; ++r)
            bufB[(32 + r) * H + h * HD + col] = (_Float16)c[r];
        }
      }
      __syncthreads();
    }

    // O-projection + residual, LN1 -> bufC (scratch role finished)
    gemm_HxH(bufB, H, 8, mats + (13 + i) * MAT_HALVES, p.bo + i * H, bufC,
             bufA, 0, wave, lane);
    __syncthreads();
    layer_norm(bufC, p.ln1g + i * H, p.ln1b + i * H, wave, lane);
    __syncthreads();

    // FFN: bufB = relu(bufC @ fw1 + fb1); bufA = LN2(bufB @ fw2 + fb2)
    gemm_HxH(bufC, H, 8, mats + (17 + i) * MAT_HALVES, p.fb1 + i * H, bufB,
             nullptr, 1, wave, lane);
    __syncthreads();
    gemm_HxH(bufB, H, 8, mats + (21 + i) * MAT_HALVES, p.fb2 + i * H, bufA,
             nullptr, 0, wave, lane);
    __syncthreads();
    layer_norm(bufA, p.ln2g + i * H, p.ln2b + i * H, wave, lane);
    __syncthreads();
  }

  // mean over tokens -> out[b][tid]
  float acc = 0.f;
  for (int r = 0; r < S; ++r) acc += (float)bufA[r * H + tid];
  p.out[b * H + tid] = acc * (1.f / 36.f);
}

extern "C" void kernel_launch(void* const* d_in, const int* in_sizes, int n_in,
                              void* d_out, int out_size, void* d_ws,
                              size_t ws_size, hipStream_t stream) {
  Params p;
  p.robot = (const float*)d_in[0];
  p.objs  = (const float*)d_in[1];
  p.masks = (const unsigned char*)d_in[2];
  p.ew1 = (const float*)d_in[3];  p.eb1 = (const float*)d_in[4];
  p.ew2 = (const float*)d_in[5];  p.eb2 = (const float*)d_in[6];
  p.elg = (const float*)d_in[7];  p.elb = (const float*)d_in[8];
  p.Wq  = (const float*)d_in[9];  p.bq  = (const float*)d_in[10];
  p.Wk  = (const float*)d_in[11]; p.bk  = (const float*)d_in[12];
  p.Wv  = (const float*)d_in[13]; p.bv  = (const float*)d_in[14];
  p.Wo  = (const float*)d_in[15]; p.bo  = (const float*)d_in[16];
  p.ln1g = (const float*)d_in[17]; p.ln1b = (const float*)d_in[18];
  p.fw1 = (const float*)d_in[19]; p.fb1 = (const float*)d_in[20];
  p.fw2 = (const float*)d_in[21]; p.fb2 = (const float*)d_in[22];
  p.ln2g = (const float*)d_in[23]; p.ln2b = (const float*)d_in[24];
  p.out = (float*)d_out;

  _Float16* ws = (_Float16*)d_ws;

  // Pass 1: pre-pack weights to f16 WMMA-fragment order in d_ws (~3.3 MB).
  int nblk = (WS_TOTAL + 255) / 256;
  prepack_kernel<<<dim3(nblk), dim3(256), 0, stream>>>(p, ws);

  // Pass 2: fused extractor, one workgroup per batch element.
  int B = in_sizes[0] / 47;  // robot_obs is [B,47]
  attn_extractor_kernel<<<dim3(B), dim3(256), 0, stream>>>(p, ws);
}
